// FaceFormerDemoV3_34445637714239
// MI455X (gfx1250) — compile-verified, matrix operand order checked by reading, not run on hardware
//
#include <hip/hip_runtime.h>
#include <hip/hip_bf16.h>
#include <math.h>

typedef __attribute__((ext_vector_type(16))) _Float16 v16h;
typedef __attribute__((ext_vector_type(8)))  float    v8f;
typedef __attribute__((ext_vector_type(2)))  _Float16 h2;

#define WMMA_F16(a,b,c) __builtin_amdgcn_wmma_f32_16x16x32_f16(false,(a),false,(b),(short)0,(c),false,false)

// Load one 16x32 f16 WMMA operand fragment for this lane from an f32 row.
// rp = pointer to the row owned by this lane (row = lane&15), h = lane>>4.
// VGPR i holds K-pair at kk = k0 + 16*(i>>2) + 8*h + 2*(i&3).
__device__ __forceinline__ v16h frag_from_row(const float* __restrict__ rp, int k0, int h){
  v16h f;
#pragma unroll
  for (int i = 0; i < 8; ++i){
    int kk = k0 + 16*(i>>2) + 8*h + 2*(i&3);
    float2 w = *reinterpret_cast<const float2*>(rp + kk);
    f[2*i]   = (_Float16)w.x;
    f[2*i+1] = (_Float16)w.y;
  }
  return f;
}

// Generic C = A @ W^T + bias (+resid) (+relu).
// One wave per 16x64 output tile: A fragment reused across 4 WMMA accumulators.
__global__ void gemm_kernel(const float* __restrict__ A, int lda,
                            const float* __restrict__ W, int ldw,
                            const float* __restrict__ bias,
                            const float* __restrict__ resid, int ldr,
                            float* __restrict__ C, int ldc,
                            int M, int N, int K, int act)
{
  int gw   = (int)((blockIdx.x * blockDim.x + threadIdx.x) >> 5);
  int lane = (int)(threadIdx.x & 31);
  int ntile = N >> 6;                       // 64-wide N blocks
  int tM = gw / ntile, tN = gw - tM * ntile;
  if (tM >= (M >> 4)) return;
  int r = lane & 15, h = lane >> 4;
  const float* arow = A + (size_t)(tM*16 + r) * lda;
  const float* wrow = W + (size_t)(tN*64 + r) * ldw;   // +c*16*ldw per sub-tile
  v8f acc[4] = {{}, {}, {}, {}};
  for (int k0 = 0; k0 < K; k0 += 32){
    v16h a = frag_from_row(arow, k0, h);
#pragma unroll
    for (int c = 0; c < 4; ++c){
      v16h b = frag_from_row(wrow + (size_t)c*16*ldw, k0, h);
      acc[c] = WMMA_F16(a, b, acc[c]);
    }
  }
#pragma unroll
  for (int c = 0; c < 4; ++c){
    int col = tN*64 + c*16 + r;
    float bv = bias[col];
#pragma unroll
    for (int q = 0; q < 8; ++q){
      int row = tM*16 + q + 8*h;
      float v = acc[c][q] + bv;
      if (resid) v += resid[(size_t)row*ldr + col];
      if (act == 1) v = fmaxf(v, 0.f);
      C[(size_t)row*ldc + col] = v;
    }
  }
}

// Conv1d (kernel 5, pad 2) over S as 5 shifted WMMA GEMMs. X: [B*S,256], Wt: [256][256][5].
// One wave per 16x64 tile. mode==1: exact GELU; mode==0: add resid.
__global__ void conv_kernel(const float* __restrict__ X,
                            const float* __restrict__ Wt,
                            const float* __restrict__ bias,
                            const float* __restrict__ resid,
                            float* __restrict__ C, int mode)
{
  int gw   = (int)((blockIdx.x * blockDim.x + threadIdx.x) >> 5);
  int lane = (int)(threadIdx.x & 31);
  const int ntile = 4;                       // N = 256 -> 4 blocks of 64
  int tM = gw / ntile, tN = gw - tM * ntile;
  if (tM >= 512) return;                     // M = 8192
  int r = lane & 15, h = lane >> 4;
  int b = (tM*16) >> 10;
  int s = ((tM*16) & 1023) + r;
  const float* wrow = Wt + (size_t)(tN*64 + r) * (256*5);
  v8f acc[4] = {{}, {}, {}, {}};
  v16h az = {};
  for (int t = 0; t < 5; ++t){
    int sp = s + t - 2;
    bool valid = (sp >= 0) && (sp < 1024);
    const float* arow = X + ((size_t)b*1024 + (size_t)(valid ? sp : 0)) * 256;
    for (int k0 = 0; k0 < 256; k0 += 32){
      v16h a = valid ? frag_from_row(arow, k0, h) : az;
#pragma unroll
      for (int c = 0; c < 4; ++c){
        const float* wr = wrow + (size_t)c*16*(256*5);
        v16h wv;
#pragma unroll
        for (int i = 0; i < 8; ++i){
          int kk = k0 + 16*(i>>2) + 8*h + 2*(i&3);
          wv[2*i]   = (_Float16)wr[(size_t)kk*5 + t];
          wv[2*i+1] = (_Float16)wr[(size_t)(kk+1)*5 + t];
        }
        acc[c] = WMMA_F16(a, wv, acc[c]);
      }
    }
  }
#pragma unroll
  for (int c = 0; c < 4; ++c){
    int col = tN*64 + c*16 + r;
    float bv = bias[col];
#pragma unroll
    for (int q = 0; q < 8; ++q){
      int row = tM*16 + q + 8*h;
      float v = acc[c][q] + bv;
      if (mode == 1) v = 0.5f * v * (1.0f + erff(v * 0.70710678118654752f));
      else           v += resid[(size_t)row*256 + col];
      C[(size_t)row*256 + col] = v;
    }
  }
}

// Motion embedding (K=33, scalar) + periodic sinusoidal PE.
__global__ void embed_kernel(const float* __restrict__ TMo,
                             const float* __restrict__ Wvm,
                             const float* __restrict__ bvm,
                             float* __restrict__ X)
{
  size_t idx = (size_t)blockIdx.x * blockDim.x + threadIdx.x;
  if (idx >= (size_t)8192*256) return;
  int d   = (int)(idx & 255);
  int row = (int)(idx >> 8);
  int b = row >> 10, t = row & 1023;
  float acc = bvm[d];
  if (t > 0){
    const float* mi = TMo + ((size_t)b*1024 + (size_t)(t-1))*33;
    const float* wr = Wvm + (size_t)d*33;
#pragma unroll
    for (int k = 0; k < 33; ++k) acc += mi[k]*wr[k];
  }
  int pos = t % 30;
  int d2  = d & ~1;
  float dv  = expf((float)d2 * (-9.210340371976184f/256.0f)); // -ln(1e4)/256
  float ang = (float)pos * dv;
  acc += (d & 1) ? cosf(ang) : sinf(ang);
  X[idx] = acc;
}

// LayerNorm over D=256, one wave32 per row.
__global__ void ln_kernel(const float* __restrict__ X,
                          const float* __restrict__ g,
                          const float* __restrict__ bt,
                          float* __restrict__ Y)
{
  int wave = (int)((blockIdx.x * blockDim.x + threadIdx.x) >> 5);
  int lane = (int)(threadIdx.x & 31);
  if (wave >= 8192) return;
  const float* x = X + (size_t)wave*256;
  float v[8]; float s = 0.f, s2 = 0.f;
#pragma unroll
  for (int i = 0; i < 8; ++i){ v[i] = x[lane + 32*i]; s += v[i]; s2 += v[i]*v[i]; }
#pragma unroll
  for (int off = 16; off; off >>= 1){ s += __shfl_xor(s, off, 32); s2 += __shfl_xor(s2, off, 32); }
  float mu  = s * (1.0f/256.0f);
  float var = s2 * (1.0f/256.0f) - mu*mu;
  float inv = rsqrtf(var + 1e-5f);
#pragma unroll
  for (int i = 0; i < 8; ++i){
    int d = lane + 32*i;
    Y[(size_t)wave*256 + d] = (v[i]-mu)*inv*g[d] + bt[d];
  }
}

__device__ __forceinline__ float mask_val(int mode, int i, int j, float slope){
  if (mode == 0){                       // causal + quantized ALiBi
    int rel = i - j;
    return (rel < 0) ? -1e30f : (-slope * (float)(rel / 30));
  }
  int dd = i - j; dd = dd < 0 ? -dd : dd; // band +/-4
  return (dd <= 4) ? 0.f : -1e30f;
}

// Flash attention: one wave per (b, head, 16-row query tile). DH=64, S=1024.
// Tile culling (exact): mode 0 -> only s0 <= t0+15; mode 1 -> band overlap only.
__global__ void attn_kernel(const float* __restrict__ Q, int ldq,
                            const float* __restrict__ Kk, int ldk,
                            const float* __restrict__ V, int ldv,
                            float* __restrict__ O, int mode)
{
  __shared__ _Float16 Pst[8][512];       // 16x32 f16 staging per wave
  int widb = (int)(threadIdx.x >> 5);
  int lane = (int)(threadIdx.x & 31);
  int wave = (int)blockIdx.x * 8 + widb;
  if (wave >= 2048) return;
  int tt = wave & 63, bh = wave >> 6;
  int b = bh >> 2, hh = bh & 3;
  int t0 = tt << 4;
  int r = lane & 15, hl = lane >> 4;

  // wave-uniform contributing key-tile range [sBeg16, sEnd16) for 16-wide tiles
  int sBeg16, sEnd16;
  if (mode == 0){ sBeg16 = 0; sEnd16 = t0 + 16; }                 // causal
  else {                                                           // band +/-4
    int lo = t0 - 19; if (lo < 0) lo = 0;
    sBeg16 = (lo >> 4) << 4;
    sEnd16 = (((t0 + 19) >> 4) + 1) << 4;
  }
  if (sEnd16 > 1024) sEnd16 = 1024;
  int sBeg32 = sBeg16 & ~31;
  int sEnd32 = (sEnd16 + 31) & ~31;
  if (sEnd32 > 1024) sEnd32 = 1024;

  const float* qrow = Q + ((size_t)b*1024 + (size_t)(t0 + r))*ldq + hh*64;
  v16h qa0 = frag_from_row(qrow,  0, hl);
  v16h qa1 = frag_from_row(qrow, 32, hl);
  const float scale = 0.125f;            // 1/sqrt(64)
  float slope = exp2f(-2.0f*(float)(hh+1)); // 0.25^(h+1)

  float mrow[8], lrow[8];
#pragma unroll
  for (int q = 0; q < 8; ++q){ mrow[q] = -1e30f; lrow[q] = 0.f; }

  const float* Kbase = Kk + ((size_t)b*1024)*ldk + hh*64;
  for (int s0 = sBeg16; s0 < sEnd16; s0 += 16){
    const float* krow = Kbase + (size_t)(s0 + r)*ldk;
    v16h kb0 = frag_from_row(krow,  0, hl);
    v16h kb1 = frag_from_row(krow, 32, hl);
    v8f sc = {};
    sc = WMMA_F16(qa0, kb0, sc);
    sc = WMMA_F16(qa1, kb1, sc);
    int j = s0 + r;
#pragma unroll
    for (int q = 0; q < 8; ++q){
      int i = t0 + q + 8*hl;
      float v  = sc[q]*scale + mask_val(mode, i, j, slope);
      float mn = fmaxf(mrow[q], v);
      lrow[q]  = lrow[q]*expf(mrow[q]-mn) + expf(v-mn);
      mrow[q]  = mn;
    }
  }
  // merge stats across the 16 lanes of each half (same rows, different cols)
#pragma unroll
  for (int off = 1; off < 16; off <<= 1){
#pragma unroll
    for (int q = 0; q < 8; ++q){
      float m2 = __shfl_xor(mrow[q], off, 32);
      float l2 = __shfl_xor(lrow[q], off, 32);
      float mn = fmaxf(mrow[q], m2);
      lrow[q]  = lrow[q]*expf(mrow[q]-mn) + l2*expf(m2-mn);
      mrow[q]  = mn;
    }
  }

  _Float16* P = &Pst[widb][0];
  const float* Vbase = V + ((size_t)b*1024)*ldv + hh*64;
  v8f o0 = {}, o1 = {}, o2 = {}, o3 = {};
  for (int s0 = sBeg32; s0 < sEnd32; s0 += 32){
#pragma unroll
    for (int hf = 0; hf < 2; ++hf){
      int ss = s0 + hf*16;
      const float* krow = Kbase + (size_t)(ss + r)*ldk;
      v16h kb0 = frag_from_row(krow,  0, hl);
      v16h kb1 = frag_from_row(krow, 32, hl);
      v8f sc = {};
      sc = WMMA_F16(qa0, kb0, sc);
      sc = WMMA_F16(qa1, kb1, sc);
      int j = ss + r;
#pragma unroll
      for (int q = 0; q < 8; ++q){
        int i = t0 + q + 8*hl;
        float v = sc[q]*scale + mask_val(mode, i, j, slope);
        P[(q + 8*hl)*32 + hf*16 + r] = (_Float16)expf(v - mrow[q]);
      }
    }
    asm volatile("s_wait_dscnt 0x0" ::: "memory"); // stores visible before A reads
    v16h pa;
#pragma unroll
    for (int i = 0; i < 8; ++i){
      int kk = 16*(i>>2) + 8*hl + 2*(i&3);
      h2 pr = *reinterpret_cast<const h2*>(&P[r*32 + kk]);
      pa[2*i] = pr[0]; pa[2*i+1] = pr[1];
    }
    asm volatile("s_wait_dscnt 0x0" ::: "memory"); // reads done before next-iter stores
#pragma unroll
    for (int c = 0; c < 4; ++c){
      v16h vb;
#pragma unroll
      for (int i = 0; i < 8; ++i){
        int kk = 16*(i>>2) + 8*hl + 2*(i&3);
        vb[2*i]   = (_Float16)Vbase[(size_t)(s0+kk  )*ldv + c*16 + r];
        vb[2*i+1] = (_Float16)Vbase[(size_t)(s0+kk+1)*ldv + c*16 + r];
      }
      if      (c==0) o0 = WMMA_F16(pa, vb, o0);
      else if (c==1) o1 = WMMA_F16(pa, vb, o1);
      else if (c==2) o2 = WMMA_F16(pa, vb, o2);
      else           o3 = WMMA_F16(pa, vb, o3);
    }
  }
  float* orow = O + ((size_t)b*1024 + (size_t)t0)*256 + hh*64;
#pragma unroll
  for (int q = 0; q < 8; ++q){
    int row = q + 8*hl;
    float inv = 1.0f / lrow[q];
    orow[(size_t)row*256 +      r] = o0[q]*inv;
    orow[(size_t)row*256 + 16 + r] = o1[q]*inv;
    orow[(size_t)row*256 + 32 + r] = o2[q]*inv;
    orow[(size_t)row*256 + 48 + r] = o3[q]*inv;
  }
}

// Final head: clip(X @ Wvr^T + bvr, 0, 1). N=33 -> scalar.
__global__ void final_kernel(const float* __restrict__ X,
                             const float* __restrict__ Wvr,
                             const float* __restrict__ bvr,
                             float* __restrict__ out)
{
  size_t idx = (size_t)blockIdx.x * blockDim.x + threadIdx.x;
  if (idx >= (size_t)8192*33) return;
  int n = (int)(idx % 33);
  size_t row = idx / 33;
  const float* x = X + row*256;
  const float* w = Wvr + (size_t)n*256;
  float acc = bvr[n];
  for (int k = 0; k < 256; ++k) acc += x[k]*w[k];
  out[idx] = fminf(fmaxf(acc, 0.f), 1.f);
}

extern "C" void kernel_launch(void* const* d_in, const int* in_sizes, int n_in,
                              void* d_out, int out_size, void* d_ws, size_t ws_size,
                              hipStream_t stream)
{
  const float* hs      = (const float*)d_in[0];
  const float* tmo     = (const float*)d_in[1];
  const float* W_af    = (const float*)d_in[2];
  const float* b_af    = (const float*)d_in[3];
  const float* c1w     = (const float*)d_in[4];
  const float* c1b     = (const float*)d_in[5];
  const float* c2w     = (const float*)d_in[6];
  const float* c2b     = (const float*)d_in[7];
  const float* Wvm     = (const float*)d_in[8];
  const float* bvm     = (const float*)d_in[9];
  const float* sa_in_w = (const float*)d_in[10];
  const float* sa_in_b = (const float*)d_in[11];
  const float* sa_out_w= (const float*)d_in[12];
  const float* sa_out_b= (const float*)d_in[13];
  const float* ca_in_w = (const float*)d_in[14];
  const float* ca_in_b = (const float*)d_in[15];
  const float* ca_out_w= (const float*)d_in[16];
  const float* ca_out_b= (const float*)d_in[17];
  const float* ln1g    = (const float*)d_in[18];
  const float* ln1b    = (const float*)d_in[19];
  const float* ln2g    = (const float*)d_in[20];
  const float* ln2b    = (const float*)d_in[21];
  const float* ln3g    = (const float*)d_in[22];
  const float* ln3b    = (const float*)d_in[23];
  const float* ff1w    = (const float*)d_in[24];
  const float* ff1b    = (const float*)d_in[25];
  const float* ff2w    = (const float*)d_in[26];
  const float* ff2b    = (const float*)d_in[27];
  const float* Wvr     = (const float*)d_in[28];
  const float* bvr     = (const float*)d_in[29];
  float* out = (float*)d_out;

  const size_t NROW = 8192;
  float* ws    = (float*)d_ws;
  float* audio = ws;                    // [8192,256]
  float* X1    = audio + NROW*256;      // [8192,256]
  float* X2    = X1    + NROW*256;      // [8192,256]
  float* Tb    = X2    + NROW*256;      // [8192,256]
  float* attno = Tb    + NROW*256;      // [8192,256]
  float* REG   = attno + NROW*256;      // [8192,768] reused: qkv / qc+kv / ffh / c1

  dim3 blk(256);
  auto gg = [](int M, int N){ int waves = (M/16)*(N/64); return dim3((unsigned)((waves*32 + 255)/256)); };

  // audio = hs @ W_af^T + b_af
  gemm_kernel<<<gg(8192,256), blk, 0, stream>>>(hs, 768, W_af, 768, b_af, nullptr, 0,
                                                audio, 256, 8192, 256, 768, 0);
  // c1 = gelu(conv(audio)) ; audio += conv(c1)
  conv_kernel<<<gg(8192,256), blk, 0, stream>>>(audio, c1w, c1b, nullptr, REG, 1);
  conv_kernel<<<gg(8192,256), blk, 0, stream>>>(REG, c2w, c2b, audio, audio, 0);
  // x = motion embed + PE
  embed_kernel<<<dim3(8192), blk, 0, stream>>>(tmo, Wvm, bvm, X1);

  for (int l = 0; l < 2; ++l){
    const float* saw = sa_in_w  + (size_t)l*768*256;
    const float* sab = sa_in_b  + (size_t)l*768;
    const float* sow = sa_out_w + (size_t)l*256*256;
    const float* sob = sa_out_b + (size_t)l*256;
    const float* caw = ca_in_w  + (size_t)l*768*256;
    const float* cab = ca_in_b  + (size_t)l*768;
    const float* cow = ca_out_w + (size_t)l*256*256;
    const float* cob = ca_out_b + (size_t)l*256;

    // self-attention: qkv (N=768), flash, out-proj + resid, LN1 -> X2
    gemm_kernel<<<gg(8192,768), blk, 0, stream>>>(X1, 256, saw, 256, sab, nullptr, 0,
                                                  REG, 768, 8192, 768, 256, 0);
    attn_kernel<<<dim3(256), blk, 0, stream>>>(REG, 768, REG+256, 768, REG+512, 768, attno, 0);
    gemm_kernel<<<gg(8192,256), blk, 0, stream>>>(attno, 256, sow, 256, sob, X1, 256,
                                                  Tb, 256, 8192, 256, 256, 0);
    ln_kernel<<<dim3(1024), blk, 0, stream>>>(Tb, ln1g + l*256, ln1b + l*256, X2);

    // cross-attention: q from X2 (N=256), kv from audio (N=512), banded flash, proj+resid, LN2 -> X1
    gemm_kernel<<<gg(8192,256), blk, 0, stream>>>(X2, 256, caw, 256, cab, nullptr, 0,
                                                  REG, 256, 8192, 256, 256, 0);
    gemm_kernel<<<gg(8192,512), blk, 0, stream>>>(audio, 256, caw + 256*256, 256, cab + 256,
                                                  nullptr, 0, REG + NROW*256, 512,
                                                  8192, 512, 256, 0);
    attn_kernel<<<dim3(256), blk, 0, stream>>>(REG, 256, REG + NROW*256, 512,
                                               REG + NROW*256 + 256, 512, attno, 1);
    gemm_kernel<<<gg(8192,256), blk, 0, stream>>>(attno, 256, cow, 256, cob, X2, 256,
                                                  Tb, 256, 8192, 256, 256, 0);
    ln_kernel<<<dim3(1024), blk, 0, stream>>>(Tb, ln2g + l*256, ln2b + l*256, X1);

    // FFN: relu(x@ff1)@ff2 + resid, LN3 -> X1
    gemm_kernel<<<gg(8192,512), blk, 0, stream>>>(X1, 256, ff1w + (size_t)l*512*256, 256,
                                                  ff1b + l*512, nullptr, 0,
                                                  REG, 512, 8192, 512, 256, 1);
    gemm_kernel<<<gg(8192,256), blk, 0, stream>>>(REG, 512, ff2w + (size_t)l*256*512, 512,
                                                  ff2b + l*256, X1, 256,
                                                  Tb, 256, 8192, 256, 512, 0);
    ln_kernel<<<dim3(1024), blk, 0, stream>>>(Tb, ln3g + l*256, ln3b + l*256, X1);
  }

  final_kernel<<<dim3((8192*33 + 255)/256), blk, 0, stream>>>(X1, Wvr, bvr, out);
}
